// FileClassifier_18013092839599
// MI455X (gfx1250) — compile-verified, matrix operand order checked by reading, not run on hardware
//
#include <hip/hip_runtime.h>
#include <hip/hip_bf16.h>

// Problem constants (match reference)
#define B_  64
#define S_  2048
#define IN_ 256
#define H_  512
#define L_  4
#define C_  2
#define NWG 16   // workgroups in the persistent recurrent kernel (one per 32 hidden cols)

typedef __attribute__((ext_vector_type(16))) __bf16 v16bf;
typedef __attribute__((ext_vector_type(8)))  float  v8f;
typedef __attribute__((ext_vector_type(4)))  int    v4i;

union FragU { v16bf v; unsigned u[8]; };
union FragH { v16bf v; __bf16  h[16]; };

// ---- CDNA5 async global->LDS copy (ASYNCcnt-tracked), with asm fallback ----
#if __has_builtin(__builtin_amdgcn_global_load_async_to_lds_b128)
#define ASYNC_COPY16(g, l)                                                     \
    __builtin_amdgcn_global_load_async_to_lds_b128(                            \
        (__attribute__((address_space(1))) v4i*)(unsigned long long)(g),       \
        (__attribute__((address_space(3))) v4i*)(unsigned)(unsigned long long)(l), \
        0, 0)
#else
#define ASYNC_COPY16(g, l)                                                     \
    asm volatile("global_load_async_to_lds_b128 %0, %1, off"                   \
                 :: "v"((unsigned)(unsigned long long)(l)),                    \
                    "v"((unsigned long long)(g)) : "memory")
#endif

#if __has_builtin(__builtin_amdgcn_s_wait_asynccnt)
#define WAIT_ASYNC(n) __builtin_amdgcn_s_wait_asynccnt(n)
#else
#define WAIT_ASYNC(n) asm volatile("s_wait_asynccnt %0" :: "i"(n) : "memory")
#endif

// K offset of the j-th packed bf16 pair for a lane-half, per CDNA5 ISA 7.12.2
// 16-bit 16x32 fragment layout: VGPR j<4 -> K 0..15 (lo/hi lane half = K 0..7 / 8..15),
// j>=4 -> K 16..31.
__device__ __forceinline__ int kpair_off(int j, int hi) {
    return (j < 4 ? 2 * j : 16 + 2 * (j - 4)) + 8 * hi;
}

// -------------------------------------------------------------------------
// outputs = softmax over a size-1 axis == all ones
__global__ void ones_kernel(float* __restrict__ out, int n) {
    int i = blockIdx.x * blockDim.x + threadIdx.x;
    if (i < n) out[i] = 1.0f;
}

__global__ void init_kernel(unsigned* __restrict__ counters) {
    if (threadIdx.x < 8) counters[threadIdx.x] = 0u;
}

// Build per-layer combined bf16 weight matrix Wc[(l,n), k] with k<512 = W_ih,
// k>=512 = W_hh (single K=1024 recurrent GEMM), plus combined bias b_ih+b_hh.
__global__ void wconv_kernel(const float* __restrict__ Wih, const float* __restrict__ Whh,
                             const float* __restrict__ bih, const float* __restrict__ bhh,
                             __bf16* __restrict__ Wcomb, float* __restrict__ bcomb) {
    long i = (long)blockIdx.x * blockDim.x + threadIdx.x;
    const long total = (long)L_ * 4 * H_ * 1024;
    if (i < total) {
        int  k  = (int)(i & 1023);
        long ln = i >> 10;                         // l*4H + n
        float v = (k < H_) ? Wih[ln * H_ + k] : Whh[ln * H_ + (k - H_)];
        Wcomb[i] = (__bf16)v;
    }
    if (i < (long)L_ * 4 * H_) bcomb[i] = bih[i] + bhh[i];
}

// -------------------------------------------------------------------------
// Input projection: xs[b,s,:] = bf16( x[b,s,:] @ lin_W^T + lin_b )
// GEMM M = B*S = 131072, N = 512, K = 256. 256 threads = 8 waves per WG,
// WG tile 128x64, wave tile 32x32 (2x2 WMMA accumulators).
__global__ void proj_kernel(const float* __restrict__ x,
                            const float* __restrict__ W,      // (H, IN) row-major
                            const float* __restrict__ bias,   // (H)
                            __bf16* __restrict__ xs) {
    const int bx   = blockIdx.x;
    const int mt   = bx >> 3;            // 1024 M-tiles of 128
    const int nt   = bx & 7;             // 8 N-tiles of 64
    const int wave = threadIdx.x >> 5;
    const int lane = threadIdx.x & 31;
    const int r    = lane & 15;
    const int hi   = lane >> 4;
    const int wr   = wave >> 1;          // 0..3
    const int wc   = wave & 1;           // 0..1
    const int row0 = mt * 128 + wr * 32;
    const int col0 = nt * 64 + wc * 32;

    v8f acc[2][2] = {};
    for (int kk = 0; kk < IN_; kk += 32) {
        FragH a[2], b[2];
        for (int rt = 0; rt < 2; ++rt) {
            const float* ap = x + (size_t)(row0 + rt * 16 + r) * IN_ + kk;
            for (int j = 0; j < 8; ++j) {
                int k0 = kpair_off(j, hi);
                a[rt].h[2 * j]     = (__bf16)ap[k0];
                a[rt].h[2 * j + 1] = (__bf16)ap[k0 + 1];
            }
        }
        for (int ct = 0; ct < 2; ++ct) {
            const float* bp = W + (size_t)(col0 + ct * 16 + r) * IN_ + kk;
            for (int j = 0; j < 8; ++j) {
                int k0 = kpair_off(j, hi);
                b[ct].h[2 * j]     = (__bf16)bp[k0];
                b[ct].h[2 * j + 1] = (__bf16)bp[k0 + 1];
            }
        }
        for (int rt = 0; rt < 2; ++rt)
            for (int ct = 0; ct < 2; ++ct)
                acc[rt][ct] = __builtin_amdgcn_wmma_f32_16x16x32_bf16(
                    false, a[rt].v, false, b[ct].v, (short)0, acc[rt][ct], false, false);
    }
    // C/D layout: VGPR e -> M = e + 8*hi, N = lane&15 (ISA 7.12.2)
    for (int rt = 0; rt < 2; ++rt)
        for (int ct = 0; ct < 2; ++ct)
            for (int e = 0; e < 8; ++e) {
                int m = row0 + rt * 16 + e + 8 * hi;
                int n = col0 + ct * 16 + (lane & 15);
                float v = acc[rt][ct][e] + bias[n];
                xs[(size_t)m * H_ + n] = (__bf16)v;
            }
}

// -------------------------------------------------------------------------
// Async-stage one timestep's activation tile xs[:, t, :] (64 x 512 bf16) into LDS.
// 4096 x b128 transfers across 256 threads (16 per thread, wave-uniform count).
__device__ __forceinline__ void stage_xs_async(const __bf16* __restrict__ xs_in, int t,
                                               __bf16* __restrict__ dstbuf, int tid) {
    for (int q = tid; q < B_ * (H_ / 8); q += 256) {
        int m = q >> 6, o = (q & 63) * 8;
        const __bf16* g = xs_in + ((size_t)m * S_ + t) * H_ + o;
        __bf16* l = dstbuf + m * H_ + o;
        ASYNC_COPY16(g, l);
    }
}

// Persistent per-layer LSTM kernel. Grid = 16 WGs x 256 threads; WG `wg` owns
// hidden columns [wg*32, wg*32+32). LDS (224KB, needs CDNA5's 320KB WGP LDS):
//   ldsX[2][64][512] bf16  double-buffered xs tile (async-prefetched, t+1 in flight)
//   ldsH[64][512]    bf16  h_prev tile (rebuilt each step, f32->bf16)
//   ldsG[4][64][32]  f32   activated gate tiles for cross-wave combine
// Per step: 8 waves = (gate 0..3, row-half 0..1), each a 32x32 WMMA tile, K=1024.
__global__ void lstm_layer_kernel(const __bf16* __restrict__ xs_in,
                                  __bf16* __restrict__ xs_out,
                                  const __bf16* __restrict__ Wc,   // (4H, 1024) bf16
                                  const float*  __restrict__ bc,   // (4H)
                                  const float*  __restrict__ h0,   // (B,H)
                                  const float*  __restrict__ c0,   // (B,H)
                                  float* __restrict__ hbuf,        // 2 * (B,H)
                                  float* __restrict__ cbuf,        // (B,H)
                                  float* __restrict__ hT,
                                  float* __restrict__ cT,
                                  unsigned* __restrict__ counter) {
    extern __shared__ char smem[];
    __bf16* ldsX = (__bf16*)smem;                 // [2][64][512]
    __bf16* ldsH = ldsX + 2 * B_ * H_;            // [64][512]
    float*  ldsG = (float*)(ldsH + B_ * H_);      // [4][64][32]

    const int wg    = blockIdx.x;
    const int tid   = threadIdx.x;
    const int wave  = tid >> 5;
    const int lane  = tid & 31;
    const int r     = lane & 15;
    const int hi    = lane >> 4;
    const int gate  = wave & 3;        // torch order: 0=i, 1=f, 2=g(tanh), 3=o
    const int rh    = wave >> 2;       // row half
    const int hbase = wg * 32;

    // prologue: kick off xs[0]
    stage_xs_async(xs_in, 0, ldsX, tid);

    for (int t = 0; t < S_; ++t) {
        const float* hread  = (t == 0) ? h0 : (hbuf + (size_t)((t + 1) & 1) * (B_ * H_));
        float*       hwrite = hbuf + (size_t)(t & 1) * (B_ * H_);

        // stage h_prev f32 -> bf16 -> ldsH
        for (int q = tid; q < B_ * (H_ / 4); q += 256) {            // 8192 x float4
            int m = q >> 7, o = (q & 127) * 4;
            float4 hv = *(const float4*)(hread + (size_t)m * H_ + o);
            __bf16* dst = ldsH + (size_t)m * H_ + o;
            dst[0] = (__bf16)hv.x; dst[1] = (__bf16)hv.y;
            dst[2] = (__bf16)hv.z; dst[3] = (__bf16)hv.w;
        }

        // kick off next step's xs tile, then wait for THIS step's tile only:
        // async loads complete in order, so asynccnt <= 16 (= the 16 just
        // issued for t+1) implies xs[t]'s 16 transfers have landed in LDS.
        if (t + 1 < S_) {
            stage_xs_async(xs_in, t + 1, ldsX + (size_t)((t + 1) & 1) * (B_ * H_), tid);
            WAIT_ASYNC(16);
        } else {
            WAIT_ASYNC(0);
        }
        __syncthreads();

        // gate GEMM: 32 rows x 32 cols per wave, K = 1024 (512 xs + 512 h)
        v8f acc[2][2] = {};
        const __bf16* Xb = ldsX + (size_t)(t & 1) * (B_ * H_);
        for (int half = 0; half < 2; ++half) {
            const __bf16* Ab = half ? ldsH : Xb;
            for (int kk = 0; kk < 512; kk += 32) {
                FragU a[2], b[2];
                for (int rt = 0; rt < 2; ++rt) {
                    const __bf16* ap = Ab + (size_t)(rh * 32 + rt * 16 + r) * 512 + kk;
                    for (int j = 0; j < 8; ++j)
                        a[rt].u[j] = *(const unsigned*)(ap + kpair_off(j, hi));
                }
                for (int ct = 0; ct < 2; ++ct) {
                    const __bf16* bp = Wc + (size_t)(gate * H_ + hbase + ct * 16 + r) * 1024
                                          + half * 512 + kk;
                    for (int j = 0; j < 8; ++j)
                        b[ct].u[j] = *(const unsigned*)(bp + kpair_off(j, hi));
                }
                for (int rt = 0; rt < 2; ++rt)
                    for (int ct = 0; ct < 2; ++ct)
                        acc[rt][ct] = __builtin_amdgcn_wmma_f32_16x16x32_bf16(
                            false, a[rt].v, false, b[ct].v, (short)0, acc[rt][ct], false, false);
            }
        }

        // bias + nonlinearity, scatter per-gate tiles to LDS
        for (int rt = 0; rt < 2; ++rt)
            for (int ct = 0; ct < 2; ++ct)
                for (int e = 0; e < 8; ++e) {
                    int m  = rh * 32 + rt * 16 + e + 8 * hi;
                    int cc = ct * 16 + (lane & 15);
                    float v = acc[rt][ct][e] + bc[gate * H_ + hbase + cc];
                    v = (gate == 2) ? tanhf(v) : 1.0f / (1.0f + __expf(-v));
                    ldsG[(gate * 64 + m) * 32 + cc] = v;
                }
        __syncthreads();

        // cell update: c columns are exclusive to this WG; h ping-pongs
        for (int e = tid; e < 64 * 32; e += 256) {
            int m = e >> 5, cc = e & 31;
            float ig = ldsG[(0 * 64 + m) * 32 + cc];
            float fg = ldsG[(1 * 64 + m) * 32 + cc];
            float gg = ldsG[(2 * 64 + m) * 32 + cc];
            float og = ldsG[(3 * 64 + m) * 32 + cc];
            size_t ci = (size_t)m * H_ + hbase + cc;
            float cold = (t == 0) ? c0[ci] : cbuf[ci];
            float cn = fg * cold + ig * gg;
            float hn = og * tanhf(cn);
            cbuf[ci]   = cn;
            hwrite[ci] = hn;
            xs_out[((size_t)m * S_ + t) * H_ + hbase + cc] = (__bf16)hn;
            if (t == S_ - 1) { hT[ci] = hn; cT[ci] = cn; }
        }

        // grid-wide barrier: all 16 WGs arrive before anyone starts step t+1
        __syncthreads();
        if (tid == 0) {
            __threadfence();
            atomicAdd(counter, 1u);
            unsigned target = (unsigned)(NWG * (t + 1));
            while (atomicAdd(counter, 0u) < target) __builtin_amdgcn_s_sleep(2);
        }
        __syncthreads();
    }
}

// -------------------------------------------------------------------------
extern "C" void kernel_launch(void* const* d_in, const int* in_sizes, int n_in,
                              void* d_out, int out_size, void* d_ws, size_t ws_size,
                              hipStream_t stream) {
    (void)in_sizes; (void)n_in; (void)out_size; (void)ws_size;
    const float* x    = (const float*)d_in[0];
    const float* linW = (const float*)d_in[1];
    const float* linb = (const float*)d_in[2];
    const float* Wih  = (const float*)d_in[3];
    const float* Whh  = (const float*)d_in[4];
    const float* bih  = (const float*)d_in[5];
    const float* bhh  = (const float*)d_in[6];
    const float* h0   = (const float*)d_in[9];
    const float* c0   = (const float*)d_in[10];
    float* out = (float*)d_out;

    // workspace layout (~286 MB)
    char* ws = (char*)d_ws;
    size_t off = 0;
    __bf16* xsA   = (__bf16*)(ws + off); off += (size_t)B_ * S_ * H_ * 2;       // 128 MB
    __bf16* xsB   = (__bf16*)(ws + off); off += (size_t)B_ * S_ * H_ * 2;       // 128 MB
    __bf16* Wcomb = (__bf16*)(ws + off); off += (size_t)L_ * 4 * H_ * 1024 * 2; // 16 MB
    float*  bcomb = (float*)(ws + off);  off += (size_t)L_ * 4 * H_ * 4;
    float*  hbuf  = (float*)(ws + off);  off += 2 * (size_t)B_ * H_ * 4;
    float*  cbuf  = (float*)(ws + off);  off += (size_t)B_ * H_ * 4;
    unsigned* counters = (unsigned*)(ws + off);

    init_kernel<<<1, 32, 0, stream>>>(counters);
    ones_kernel<<<(B_ * S_ * C_ + 255) / 256, 256, 0, stream>>>(out, B_ * S_ * C_);

    long wtot = (long)L_ * 4 * H_ * 1024;
    wconv_kernel<<<(int)((wtot + 255) / 256), 256, 0, stream>>>(Wih, Whh, bih, bhh, Wcomb, bcomb);

    proj_kernel<<<(B_ * S_ / 128) * (H_ / 64), 256, 0, stream>>>(x, linW, linb, xsA);

    float* hT = out + B_ * S_ * C_;
    float* cT = hT + (size_t)L_ * B_ * H_;
    size_t smem = (size_t)(2 * B_ * H_ + B_ * H_) * sizeof(__bf16)   // ldsX + ldsH
                + (size_t)4 * 64 * 32 * sizeof(float);               // ldsG
    for (int l = 0; l < L_; ++l) {
        const __bf16* xin = (l & 1) ? xsB : xsA;
        __bf16*      xout = (l & 1) ? xsA : xsB;
        lstm_layer_kernel<<<NWG, 256, smem, stream>>>(
            xin, xout,
            Wcomb + (size_t)l * 4 * H_ * 1024, bcomb + (size_t)l * 4 * H_,
            h0 + (size_t)l * B_ * H_, c0 + (size_t)l * B_ * H_,
            hbuf, cbuf,
            hT + (size_t)l * B_ * H_, cT + (size_t)l * B_ * H_,
            counters + l);
    }
}
